// AccumulatorFunction_55413668053098
// MI455X (gfx1250) — matrix operand order repaired and checked by reading, not compiled, not run on hardware
//
#include <hip/hip_runtime.h>
#include <hip/hip_bf16.h>
#include <stdint.h>

// MI455X / gfx1250 fused neuron-attention:
//   q,k,v = x@W^T (WMMA f16->f32), rel = adj^T .* (q@k^T),
//   out   = (rel @ v) * rsqrt(rowsum(rel^2))
// L2 normalization is a linear post-scale, so keys are streamed
// flash-attention style: rel never touches HBM.
//
// Key-split layout: workgroup = 4 waves per (batch, 16-query tile).
//  * Q tile (16x512 f16, 16KB) staged to LDS via global_load_async_to_lds_b128.
//  * Wave w computes full-D QK^T for key block w of each 128-key step,
//    masks + accumulates sumsq locally, publishes masked rel (f16) to LDS.
//  * Each wave then consumes all 4 rel blocks against its private 128-wide
//    D-chunk of V (stored transposed by the projection kernel) -> 64 fp32
//    accumulator registers per wave, no cross-wave f32 reductions.

typedef __attribute__((ext_vector_type(16))) _Float16 v16h;
typedef __attribute__((ext_vector_type(8)))  _Float16 v8h;
typedef __attribute__((ext_vector_type(8)))  float    v8f;

#define BB 8
#define NN 2048
#define DD 512

// ---------------------------------------------------------------------------
// fp32 -> f16 operand packing helper: 8 consecutive floats into dst[base..base+7]
// ---------------------------------------------------------------------------
__device__ __forceinline__ void cvt8(v16h& dst, int base, const float* __restrict__ p) {
    float4 u = *(const float4*)p;
    float4 v = *(const float4*)(p + 4);
    dst[base + 0] = (_Float16)u.x; dst[base + 1] = (_Float16)u.y;
    dst[base + 2] = (_Float16)u.z; dst[base + 3] = (_Float16)u.w;
    dst[base + 4] = (_Float16)v.x; dst[base + 5] = (_Float16)v.y;
    dst[base + 6] = (_Float16)v.z; dst[base + 7] = (_Float16)v.w;
}

// ---------------------------------------------------------------------------
// Kernel 1: projections. One wave32 per 16x16 output tile.
// y = x@W^T => both operands contiguous along d. V stored transposed [B][D][N].
// ---------------------------------------------------------------------------
__global__ __launch_bounds__(32) void qkv_proj_kernel(
    const float* __restrict__ x,
    const float* __restrict__ Wq,
    const float* __restrict__ Wk,
    const float* __restrict__ Wv,
    _Float16* __restrict__ Qh,      // [B*N, D]
    _Float16* __restrict__ Kh,      // [B*N, D]
    _Float16* __restrict__ Vt)      // [B][D][N]
{
    const int lane  = threadIdx.x;
    const int half  = lane >> 4;
    const int lm    = lane & 15;
    const int rowT  = blockIdx.x;
    const int colT  = blockIdx.y;
    const int which = blockIdx.z;

    const float* W = (which == 0) ? Wq : (which == 1) ? Wk : Wv;
    const int row  = rowT * 16 + lm;
    const int ncol = colT * 16 + lm;

    v8f acc = {};
    for (int kb = 0; kb < DD; kb += 32) {
        v16h a;   // A-frag: lane row=l%16, K = kb+(l/16)*8+{0..7} / +16
        const float* ap = x + (size_t)row * DD + kb + half * 8;
        cvt8(a, 0, ap);
        cvt8(a, 8, ap + 16);

        v16h bm;  // B-frag: lane col n=l%16, K = kb+(l/16)*8+{0..7} / +16; B[k][n]=W[n][k]
        const float* bp = W + (size_t)ncol * DD + kb + half * 8;
        cvt8(bm, 0, bp);
        cvt8(bm, 8, bp + 16);

        acc = __builtin_amdgcn_wmma_f32_16x16x32_f16(
            false, a, false, bm, (short)0, acc, false, false);
    }

    if (which < 2) {
        _Float16* dst = (which == 0) ? Qh : Kh;
#pragma unroll
        for (int r = 0; r < 8; ++r) {
            const int orow = rowT * 16 + r + half * 8;
            dst[(size_t)orow * DD + ncol] = (_Float16)acc[r];
        }
    } else {
        const int b0 = (rowT * 16) / NN;
#pragma unroll
        for (int r = 0; r < 8; ++r) {
            const int orow = rowT * 16 + r + half * 8;
            Vt[((size_t)b0 * DD + ncol) * NN + (orow - b0 * NN)] = (_Float16)acc[r];
        }
    }
}

// ---------------------------------------------------------------------------
// Kernel 2: fused masked attention with L2 row-normalization.
// ---------------------------------------------------------------------------
__global__ __launch_bounds__(128) void fused_attn_kernel(
    const _Float16* __restrict__ Qh,    // [B*N, D]
    const _Float16* __restrict__ Kh,    // [B*N, D]
    const _Float16* __restrict__ Vt,    // [B][D][N]
    const float*    __restrict__ adj,   // [N, N]
    float*          __restrict__ out)   // [B, N, D]
{
    __shared__ _Float16 Qs[16][DD];        // 16 KB: staged Q tile
    __shared__ _Float16 arel[4][16][32];   // 4 KB : masked rel, f16, per key block
    __shared__ float    ssred[4][16];      // per-wave partial sumsq

    const int tid  = threadIdx.x;
    const int w    = tid >> 5;
    const int lane = tid & 31;
    const int half = lane >> 4;
    const int lm   = lane & 15;
    const int b    = blockIdx.x;
    const int qt   = blockIdx.y;
    const int dch  = w * 128;              // this wave's output D-chunk

    const _Float16* K = Kh + (size_t)b * NN * DD;
    const _Float16* V = Vt + (size_t)b * DD * NN;

    // ---- stage Q tile (16 rows x 512, contiguous 16KB) via async copy to LDS
    {
        typedef __attribute__((address_space(3))) _Float16 lds_f16;
        const uint32_t lds_base =
            (uint32_t)(uintptr_t)(lds_f16*)&Qs[0][0];
        const char* gsrc = (const char*)(Qh + (size_t)(b * NN + qt * 16) * DD);
#pragma unroll
        for (int i = 0; i < 8; ++i) {
            const uint32_t off = (uint32_t)(i * 128 + tid) * 16u;   // 16B per lane
            asm volatile("global_load_async_to_lds_b128 %0, %1, off"
                         :: "v"(lds_base + off), "v"(gsrc + off)
                         : "memory");
        }
        asm volatile("s_wait_asynccnt 0" ::: "memory");
    }
    __syncthreads();

    v8f outAcc[8];
#pragma unroll
    for (int t = 0; t < 8; ++t) outAcc[t] = (v8f){};
    float ss[8] = {0.f, 0.f, 0.f, 0.f, 0.f, 0.f, 0.f, 0.f};

    for (int mb0 = 0; mb0 < NN; mb0 += 128) {
        const int kb0 = mb0 + w * 32;       // this wave's 32-key block

        // ---- full-D QK^T for this wave's keys: 2 sub-tiles x 16 K-steps
        v8f rel0 = {}, rel1 = {};
#pragma unroll 4
        for (int s = 0; s < 16; ++s) {
            // shared A-frag from LDS Q: lane row m=l%16, K=(l/16)*16+{0..15}
            v8h qlo = *(const v8h*)&Qs[lm][s * 32 + half * 16];
            v8h qhi = *(const v8h*)&Qs[lm][s * 32 + half * 16 + 8];
            v16h af;
#pragma unroll
            for (int i = 0; i < 8; ++i) { af[i] = qlo[i]; af[8 + i] = qhi[i]; }

            // B-frags: B[d][n] = K[key n][d], contiguous 32B along d
            v16h kf0 = *(const v16h*)(K + (size_t)(kb0 + lm)      * DD + s * 32 + half * 16);
            v16h kf1 = *(const v16h*)(K + (size_t)(kb0 + 16 + lm) * DD + s * 32 + half * 16);

            rel0 = __builtin_amdgcn_wmma_f32_16x16x32_f16(
                false, af, false, kf0, (short)0, rel0, false, false);
            rel1 = __builtin_amdgcn_wmma_f32_16x16x32_f16(
                false, af, false, kf1, (short)0, rel1, false, false);
        }

        // ---- mask by adj^T, accumulate sumsq, publish f16 rel block
        // C layout: lane l, vgpr r -> q-row i = r+8*(l/16), key col j = l%16.
        // mask(i,j) = adj[j][i] -> 8 contiguous floats per lane per sub-tile.
#pragma unroll
        for (int sub = 0; sub < 2; ++sub) {
            const int key = kb0 + sub * 16 + lm;
            const float* adjp = adj + (size_t)key * NN + qt * 16 + half * 8;
            float4 m0 = *(const float4*)adjp;
            float4 m1 = *(const float4*)(adjp + 4);
            float msk[8] = {m0.x, m0.y, m0.z, m0.w, m1.x, m1.y, m1.z, m1.w};
#pragma unroll
            for (int r = 0; r < 8; ++r) {
                float v = ((sub == 0) ? rel0[r] : rel1[r]) * msk[r];
                ss[r] += v * v;
                arel[w][r + half * 8][sub * 16 + lm] = (_Float16)v;
            }
        }
        __syncthreads();

        // ---- out(D-chunk) += sum over 4 key blocks: rel(16x32) @ V(32x128)
#pragma unroll
        for (int blk = 0; blk < 4; ++blk) {
            v8h rlo = *(const v8h*)&arel[blk][lm][half * 16];
            v8h rhi = *(const v8h*)&arel[blk][lm][half * 16 + 8];
            v16h af;
#pragma unroll
            for (int i = 0; i < 8; ++i) { af[i] = rlo[i]; af[8 + i] = rhi[i]; }

            const int keyb = mb0 + blk * 32;
#pragma unroll
            for (int t = 0; t < 8; ++t) {
                // B[k][n] = Vt[b][dch+t*16+n][keyb+k], contiguous 32B along k
                v16h vf = *(const v16h*)(V + (size_t)(dch + t * 16 + lm) * NN
                                           + keyb + half * 16);
                outAcc[t] = __builtin_amdgcn_wmma_f32_16x16x32_f16(
                    false, af, false, vf, (short)0, outAcc[t], false, false);
            }
        }
        __syncthreads();   // arel reused next iteration
    }

    // ---- sumsq: butterfly within the 16-lane group owning each row half,
    //      then cross-wave add through LDS.
#pragma unroll
    for (int r = 0; r < 8; ++r) {
        float s = ss[r];
        s += __shfl_xor(s, 1);
        s += __shfl_xor(s, 2);
        s += __shfl_xor(s, 4);
        s += __shfl_xor(s, 8);
        ss[r] = s;
    }
    if (lm == 0) {
#pragma unroll
        for (int r = 0; r < 8; ++r) ssred[w][r + half * 8] = ss[r];
    }
    __syncthreads();

    float inv[8];
#pragma unroll
    for (int r = 0; r < 8; ++r) {
        const int row = r + half * 8;
        inv[r] = rsqrtf(ssred[0][row] + ssred[1][row] +
                        ssred[2][row] + ssred[3][row]);
    }

    // ---- scaled fp32 stores
#pragma unroll
    for (int t = 0; t < 8; ++t) {
#pragma unroll
        for (int r = 0; r < 8; ++r) {
            const int orow = qt * 16 + r + half * 8;
            out[((size_t)b * NN + orow) * DD + dch + t * 16 + lm] =
                outAcc[t][r] * inv[r];
        }
    }
}

// ---------------------------------------------------------------------------
extern "C" void kernel_launch(void* const* d_in, const int* in_sizes, int n_in,
                              void* d_out, int out_size, void* d_ws, size_t ws_size,
                              hipStream_t stream) {
    (void)in_sizes; (void)n_in; (void)out_size; (void)ws_size;
    const float* x   = (const float*)d_in[0];
    const float* adj = (const float*)d_in[1];
    const float* Wq  = (const float*)d_in[2];
    const float* Wk  = (const float*)d_in[3];
    const float* Wv  = (const float*)d_in[4];
    float* outp      = (float*)d_out;

    const size_t elems = (size_t)BB * NN * DD;
    _Float16* Qh = (_Float16*)d_ws;
    _Float16* Kh = Qh + elems;
    _Float16* Vt = Kh + elems;

    dim3 g1(BB * NN / 16, DD / 16, 3);
    qkv_proj_kernel<<<g1, 32, 0, stream>>>(x, Wq, Wk, Wv, Qh, Kh, Vt);

    dim3 g2(BB, NN / 16);
    fused_attn_kernel<<<g2, 128, 0, stream>>>(Qh, Kh, Vt, adj, outp);
}
